// ScatTransform1D_46523085750861
// MI455X (gfx1250) — matrix operand order repaired and checked by reading, not compiled, not run on hardware
//
#include <hip/hip_runtime.h>
#include <math.h>

typedef __attribute__((ext_vector_type(16))) _Float16 v16h;
typedef __attribute__((ext_vector_type(2)))  _Float16 h2;
typedef __attribute__((ext_vector_type(8)))  float    v8f;
// Filter rows are only dword-aligned (Lmax is odd) -> 4-byte-aligned vectors.
typedef float f4u __attribute__((ext_vector_type(4), aligned(4)));

#define TTILE_FAST 256   // fast path: 8 waves x (2 x 16) time samples
#define TTILE_FB   128   // fallback:  8 waves x 16 time samples

#define WMMA_F16(A, B, C) \
    __builtin_amdgcn_wmma_f32_16x16x32_f16(false, (A), false, (B), (short)0, (C), false, false)

// ---------------------------------------------------------------------------
// Pre-swizzle pass: convert the f32 filter bank into f16 A-fragments stored
// in WMMA register order. Fragment for (chunk, lane) = 16 contiguous halves:
//   frag[chunk][lane][h] = filt[k0 + (lane&15)]
//                              [l_lo + chunk*32 + (h&7) + (h>=8?16:0) + (lane>>4)*8]
// OOB taps (>= Lmax) are stored as zero, so the GEMM loop needs no guards.
// ---------------------------------------------------------------------------
__global__ __launch_bounds__(256) void scat1d_swizzle_kernel(
    const float* __restrict__ fr, const float* __restrict__ fi,
    _Float16* __restrict__ wsr, _Float16* __restrict__ wsi,
    int Lmax, int k0, int l_lo, int chunkBase)
{
    const int c  = blockIdx.x;
    const int f0 = threadIdx.x * 2;          // halves f0, f0+1 (same lane)
    const int lane = f0 >> 4;
    const int h0   = f0 & 15;                // even
    const int n  = lane & 15;
    const int hg = lane >> 4;

    const float* rr = fr + (size_t)(k0 + n) * Lmax;
    const float* ri = fi + (size_t)(k0 + n) * Lmax;

    const int kk0 = (h0 & 7) + ((h0 >= 8) ? 16 : 0) + hg * 8;
    const int i0  = l_lo + c * 32 + kk0;
    const int i1  = i0 + 1;                  // h0 even -> consecutive taps

    int c0 = i0 < Lmax ? i0 : Lmax - 1;
    int c1 = i1 < Lmax ? i1 : Lmax - 1;
    float r0 = rr[c0]; r0 = (i0 < Lmax) ? r0 : 0.0f;
    float r1 = rr[c1]; r1 = (i1 < Lmax) ? r1 : 0.0f;
    float q0 = ri[c0]; q0 = (i0 < Lmax) ? q0 : 0.0f;
    float q1 = ri[c1]; q1 = (i1 < Lmax) ? q1 : 0.0f;

    const size_t base = (size_t)(chunkBase + c) * 512 + (size_t)f0;
    *(h2*)(wsr + base) = __builtin_bit_cast(h2, __builtin_amdgcn_cvt_pkrtz(r0, r1));
    *(h2*)(wsi + base) = __builtin_bit_cast(h2, __builtin_amdgcn_cvt_pkrtz(q0, q1));
}

// ---------------------------------------------------------------------------
// Fast path: 16 filters x 256 times per block; wave w owns two 16x16 C tiles
// (real+imag each) at times t0+32w and t0+32w+16. Per 32-tap chunk:
//   A: 4 x global_load_b128 (pre-swizzled f16 fragments, no converts)
//   B: 4 x ds_load_b128     (f16 x-window in LDS)
//   4 x v_wmma_f32_16x16x32_f16
// ---------------------------------------------------------------------------
__global__ __launch_bounds__(256) void scat1d_wmma_fast_kernel(
    const float* __restrict__ x,        // (B, T) f32
    const _Float16* __restrict__ wsr,   // swizzled real fragments
    const _Float16* __restrict__ wsi,   // swizzled imag fragments
    float* __restrict__ out,            // (B, K, T) f32
    int T, int K, int P,
    int k0, int l_lo, int nch, int chunkBase)
{
    extern __shared__ _Float16 xwin[];

    const int tid  = threadIdx.x;
    const int lane = tid & 31;
    const int wave = tid >> 5;
    const int b    = blockIdx.y;
    const int t0   = blockIdx.x * TTILE_FAST;

    // Stage x window into LDS (f16), branchless.
    const int wlen = nch * 32 + 288;
    const float* xb = x + (size_t)b * T;
    for (int i = tid; i < wlen; i += 256) {
        int g  = t0 + l_lo + i - P;
        int gc = g < 0 ? 0 : (g >= T ? T - 1 : g);
        float v = xb[gc];
        v = ((unsigned)g < (unsigned)T) ? v : 0.0f;
        xwin[i] = (_Float16)v;
    }
    __syncthreads();

    const int n  = lane & 15;
    const int hg = lane >> 4;

    const _Float16* fragr = wsr + (size_t)chunkBase * 512 + (size_t)lane * 16;
    const _Float16* fragi = wsi + (size_t)chunkBase * 512 + (size_t)lane * 16;

    // B fragment (32x16 f16): col n = lane&15; half h -> tap (h&7) + (h>=8?16:0) + hg*8.
    const _Float16* wb0 = xwin + wave * 32 + n + hg * 8;
    const _Float16* wb1 = wb0 + 16;

    v8f cr0 = {}, ci0 = {}, cr1 = {}, ci1 = {};

#pragma unroll 2
    for (int c = 0; c < nch; ++c) {
        v16h ar = *(const v16h*)(fragr + (size_t)c * 512);
        v16h ai = *(const v16h*)(fragi + (size_t)c * 512);

        v16h b0, b1;
        const _Float16* wp0 = wb0 + c * 32;
        const _Float16* wp1 = wb1 + c * 32;
#pragma unroll
        for (int h = 0; h < 16; ++h) { b0[h] = wp0[(h & 7) + ((h >= 8) ? 16 : 0)]; b1[h] = wp1[(h & 7) + ((h >= 8) ? 16 : 0)]; }

        cr0 = WMMA_F16(ar, b0, cr0);
        ci0 = WMMA_F16(ai, b0, ci0);
        cr1 = WMMA_F16(ar, b1, cr1);
        ci1 = WMMA_F16(ai, b1, ci1);
    }

    // Modulus + store both time sub-tiles.
    const int t    = t0 + wave * 32 + n;
    const size_t ob = ((size_t)b * K + (size_t)(k0 + hg * 8)) * (size_t)T + (size_t)t;
#pragma unroll
    for (int r = 0; r < 8; ++r) {
        float vr0 = cr0[r], vi0 = ci0[r];
        float vr1 = cr1[r], vi1 = ci1[r];
        out[ob + (size_t)r * T]      = sqrtf(vr0 * vr0 + vi0 * vi0);
        out[ob + (size_t)r * T + 16] = sqrtf(vr1 * vr1 + vi1 * vi1);
    }
}

// ---------------------------------------------------------------------------
// Fallback (no workspace): direct f32 filter loads + packed converts.
// ---------------------------------------------------------------------------
__device__ __forceinline__ void cvt8(v16h& a, int base, f4u lo, f4u hi) {
    h2 p;
    p = __builtin_bit_cast(h2, __builtin_amdgcn_cvt_pkrtz(lo.x, lo.y));
    a[base + 0] = p.x; a[base + 1] = p.y;
    p = __builtin_bit_cast(h2, __builtin_amdgcn_cvt_pkrtz(lo.z, lo.w));
    a[base + 2] = p.x; a[base + 3] = p.y;
    p = __builtin_bit_cast(h2, __builtin_amdgcn_cvt_pkrtz(hi.x, hi.y));
    a[base + 4] = p.x; a[base + 5] = p.y;
    p = __builtin_bit_cast(h2, __builtin_amdgcn_cvt_pkrtz(hi.z, hi.w));
    a[base + 6] = p.x; a[base + 7] = p.y;
}

__global__ __launch_bounds__(256) void scat1d_wmma_fb_kernel(
    const float* __restrict__ x,
    const float* __restrict__ fr,
    const float* __restrict__ fi,
    float* __restrict__ out,
    int T, int K, int Lmax, int P,
    int k0, int l_lo, int nch)
{
    extern __shared__ _Float16 xwin[];

    const int tid  = threadIdx.x;
    const int lane = tid & 31;
    const int wave = tid >> 5;
    const int b    = blockIdx.y;
    const int t0   = blockIdx.x * TTILE_FB;

    const int wlen = nch * 32 + 160;
    const float* xb = x + (size_t)b * T;
    for (int i = tid; i < wlen; i += 256) {
        int g  = t0 + l_lo + i - P;
        int gc = g < 0 ? 0 : (g >= T ? T - 1 : g);
        float v = xb[gc];
        v = ((unsigned)g < (unsigned)T) ? v : 0.0f;
        xwin[i] = (_Float16)v;
    }
    __syncthreads();

    const int n  = lane & 15;
    const int hg = lane >> 4;

    const float* rowr = fr + (size_t)(k0 + n) * Lmax;
    const float* rowi = fi + (size_t)(k0 + n) * Lmax;
    const _Float16* wb = xwin + wave * 16 + n + hg * 8;

    v8f cr = {};
    v8f ci = {};

#pragma unroll 2
    for (int c = 0; c < nch - 1; ++c) {
        const int l = l_lo + c * 32;
        const float* pr = rowr + l + hg * 8;
        const float* pi = rowi + l + hg * 8;

        f4u r0 = *(const f4u*)(pr);
        f4u r1 = *(const f4u*)(pr + 4);
        f4u r2 = *(const f4u*)(pr + 16);
        f4u r3 = *(const f4u*)(pr + 20);
        f4u q0 = *(const f4u*)(pi);
        f4u q1 = *(const f4u*)(pi + 4);
        f4u q2 = *(const f4u*)(pi + 16);
        f4u q3 = *(const f4u*)(pi + 20);

        v16h ar, ai;
        cvt8(ar, 0, r0, r1); cvt8(ar, 8, r2, r3);
        cvt8(ai, 0, q0, q1); cvt8(ai, 8, q2, q3);

        v16h bx;
        const _Float16* wp = wb + c * 32;
#pragma unroll
        for (int h = 0; h < 16; ++h) bx[h] = wp[(h & 7) + ((h >= 8) ? 16 : 0)];

        cr = WMMA_F16(ar, bx, cr);
        ci = WMMA_F16(ai, bx, ci);
    }

    if (nch > 0) {
        const int l = l_lo + (nch - 1) * 32;
        v16h ar, ai;
#pragma unroll
        for (int h = 0; h < 16; ++h) {
            int kk  = (h & 7) + ((h >= 8) ? 16 : 0) + hg * 8;
            int idx = l + kk;
            int idc = idx < Lmax ? idx : Lmax - 1;
            float r = rowr[idc];
            float q = rowi[idc];
            bool ok = idx < Lmax;
            ar[h] = (_Float16)(ok ? r : 0.0f);
            ai[h] = (_Float16)(ok ? q : 0.0f);
        }
        v16h bx;
        const _Float16* wp = wb + (nch - 1) * 32;
#pragma unroll
        for (int h = 0; h < 16; ++h) bx[h] = wp[(h & 7) + ((h >= 8) ? 16 : 0)];

        cr = WMMA_F16(ar, bx, cr);
        ci = WMMA_F16(ai, bx, ci);
    }

    const int t = t0 + wave * 16 + n;
    const size_t ob = ((size_t)b * K + (size_t)(k0 + hg * 8)) * (size_t)T + (size_t)t;
#pragma unroll
    for (int r = 0; r < 8; ++r) {
        float vr = cr[r];
        float vi = ci[r];
        out[ob + (size_t)r * T] = sqrtf(vr * vr + vi * vi);
    }
}

// ---------------------------------------------------------------------------
extern "C" void kernel_launch(void* const* d_in, const int* in_sizes, int n_in,
                              void* d_out, int out_size, void* d_ws, size_t ws_size,
                              hipStream_t stream)
{
    const float* x  = (const float*)d_in[0];   // (2, 1, 32768) f32
    const float* fr = (const float*)d_in[1];   // (192, Lmax) f32
    const float* fi = (const float*)d_in[2];   // (192, Lmax) f32
    float* out = (float*)d_out;                // (2, 192, 32768) f32

    const int B = 2, K = 192, T = 32768, J = 12;
    const int Lmax = in_sizes[1] / K;          // padded filter length (odd)
    const int P    = Lmax / 2;                 // center index / pad amount

    // Per-octave tap ranges (filters j*16 .. j*16+15 share support within ~2x).
    int l_lo[12], nch[12], cbase[12];
    int tot = 0;
    for (int j = 0; j < J; ++j) {
        int M  = (int)ceil(6.0 * pow(2.0, (double)j + 15.0 / 16.0)) + 4;
        int lo = P - M;
        if (lo < 0) lo = 0;
        lo &= ~31;                             // chunk-align (stays >= 0)
        int hi = P + M;
        if (hi > Lmax - 1) hi = Lmax - 1;
        l_lo[j]  = lo;
        nch[j]   = (hi - lo + 32) / 32;
        cbase[j] = tot;
        tot += nch[j];
    }

    const size_t needWs = (size_t)tot * 1024 * 2;   // r+i, 512 halves each, 2B
    if (ws_size >= needWs) {
        _Float16* wsr = (_Float16*)d_ws;
        _Float16* wsi = wsr + (size_t)tot * 512;

        for (int j = 0; j < J; ++j)
            scat1d_swizzle_kernel<<<dim3(nch[j]), 256, 0, stream>>>(
                fr, fi, wsr, wsi, Lmax, j * 16, l_lo[j], cbase[j]);

        dim3 grid(T / TTILE_FAST, B);
        for (int j = 0; j < J; ++j) {
            size_t smem = (size_t)(nch[j] * 32 + 288) * 2;
            scat1d_wmma_fast_kernel<<<grid, 256, smem, stream>>>(
                x, wsr, wsi, out, T, K, P, j * 16, l_lo[j], nch[j], cbase[j]);
        }
    } else {
        dim3 grid(T / TTILE_FB, B);
        for (int j = 0; j < J; ++j) {
            size_t smem = (size_t)(nch[j] * 32 + 160) * 2;
            scat1d_wmma_fb_kernel<<<grid, 256, smem, stream>>>(
                x, fr, fi, out, T, K, Lmax, P, j * 16, l_lo[j], nch[j]);
        }
    }
}
